// FlashInfer_NormQKNorm_317827580193
// MI455X (gfx1250) — matrix operand order repaired and checked by reading, not compiled, not run on hardware
//
#include <hip/hip_runtime.h>
#include <hip/hip_bf16.h>
#include <stdint.h>

typedef __attribute__((ext_vector_type(16))) __bf16 v16bf;
typedef __attribute__((ext_vector_type(8)))  float  v8f;

#define M_ 1024
#define N_ 4096
#define D_ 128
#define H_ 32
#define P_ 4096
#define L_ 5120  /* P_ + M_ */

union AF { v16bf v; uint4 u[2]; };

// CDNA5 async global->LDS copy (ASYNCcnt). INST_OFFSET applies to BOTH the
// LDS destination and the global source address, so one (lds,addr) pair plus
// offsets covers a contiguous row.
#define ASYNC_LD128(lds, gp, OFF)                                         \
  asm volatile("global_load_async_to_lds_b128 %0, %1, off offset:" #OFF   \
               :: "v"(lds), "v"(gp) : "memory")
#define WAIT_ASYNCCNT0() asm volatile("s_wait_asynccnt 0x0" ::: "memory")

static __device__ __forceinline__ uint32_t lds_addr32(const void* p) {
  // flat LDS addresses keep the wave-relative LDS offset in bits [31:0]
  return (uint32_t)(uintptr_t)p;
}

// ---------------------------------------------------------------- RMSNorm(X)
__global__ void __launch_bounds__(256) k_rmsnorm_x(const float* __restrict__ X,
                                                   __bf16* __restrict__ Xn) {
  __shared__ float red[256];
  int row = blockIdx.x;
  int t = threadIdx.x;
  const float* xr = X + (size_t)row * N_;
  float vals[16];
  float ssq = 0.f;
  for (int i = 0; i < 16; ++i) { float v = xr[t + i * 256]; vals[i] = v; ssq += v * v; }
  red[t] = ssq; __syncthreads();
  for (int s = 128; s > 0; s >>= 1) { if (t < s) red[t] += red[t + s]; __syncthreads(); }
  float r = rsqrtf(red[0] * (1.0f / N_));
  __bf16* o = Xn + (size_t)row * N_;
  for (int i = 0; i < 16; ++i) o[t + i * 256] = (__bf16)(vals[i] * r);
}

// ------------------------------------------- W (f32) -> Wt (bf16, transposed)
__global__ void __launch_bounds__(256) k_wt(const float* __restrict__ W0,
                                            const float* __restrict__ W1,
                                            const float* __restrict__ W2,
                                            __bf16* __restrict__ WtAll) {
  __shared__ float tile[32][33];
  const float* W = blockIdx.z == 0 ? W0 : (blockIdx.z == 1 ? W1 : W2);
  __bf16* out = WtAll + (size_t)blockIdx.z * N_ * N_;
  int k0 = blockIdx.x * 32, j0 = blockIdx.y * 32;
  int lx = threadIdx.x & 31, ly = threadIdx.x >> 5;  // ly in 0..7
  for (int s = 0; s < 4; ++s) {
    int r = ly + s * 8;
    tile[r][lx] = W[(size_t)(k0 + r) * N_ + j0 + lx];
  }
  __syncthreads();
  for (int s = 0; s < 4; ++s) {
    int r = ly + s * 8;
    out[(size_t)(j0 + r) * N_ + k0 + lx] = (__bf16)tile[lx][r];
  }
}

// ------------------------------------------------- cache_K (f32) -> Kc (bf16)
__global__ void __launch_bounds__(256) k_cachek(const float* __restrict__ cK,
                                                __bf16* __restrict__ Kc) {
  size_t stride = (size_t)gridDim.x * blockDim.x;
  size_t tot = (size_t)H_ * P_ * D_;
  for (size_t i = (size_t)blockIdx.x * blockDim.x + threadIdx.x; i < tot; i += stride) {
    size_t h = i / ((size_t)P_ * D_);
    size_t rem = i - h * ((size_t)P_ * D_);      // n*D + d, n < P
    Kc[(h * L_) * D_ + rem] = (__bf16)cK[i];
  }
}

// ---------------------------------- cache_V (f32) -> Vt (bf16, [h][d][L] T)
__global__ void __launch_bounds__(256) k_cachev(const float* __restrict__ cV,
                                                __bf16* __restrict__ Vt) {
  __shared__ float tile[32][33];
  int h = blockIdx.z, n0 = blockIdx.x * 32, d0 = blockIdx.y * 32;
  int lx = threadIdx.x & 31, ly = threadIdx.x >> 5;
  for (int s = 0; s < 4; ++s) {
    int n = ly + s * 8;
    tile[n][lx] = cV[((size_t)h * P_ + n0 + n) * D_ + d0 + lx];
  }
  __syncthreads();
  for (int s = 0; s < 4; ++s) {
    int d = ly + s * 8;
    Vt[((size_t)h * D_ + d0 + d) * L_ + n0 + lx] = (__bf16)tile[lx][d];
  }
}

// ------------------- bf16 WMMA GEMM (q/k/v), async double-buffered staging
#define AP 40  // LDS row pitch in bf16 (80B: 16B-aligned, 20-dword bank stride)
__global__ void __launch_bounds__(256) k_gemm(const __bf16* __restrict__ Xn,
                                              const __bf16* __restrict__ WtAll,
                                              float* __restrict__ Qf,
                                              float* __restrict__ Kf,
                                              float* __restrict__ Vf) {
  __shared__ __attribute__((aligned(16))) __bf16 As[2][128 * AP];
  __shared__ __attribute__((aligned(16))) __bf16 Bs[2][128 * AP];
  const __bf16* Wt = WtAll + (size_t)blockIdx.z * N_ * N_;
  float* Out = blockIdx.z == 0 ? Qf : (blockIdx.z == 1 ? Kf : Vf);
  int c0 = blockIdx.x * 128, m0 = blockIdx.y * 128;
  int t = threadIdx.x;
  int lane = t & 31, w = t >> 5;
  int rw = (w & 3) * 32, cw = (w >> 2) * 64;
  int ln = lane & 15, hh = lane >> 4;
  v8f zero = {};
  v8f acc[2][4];
  for (int rt = 0; rt < 2; ++rt)
    for (int ct = 0; ct < 4; ++ct) acc[rt][ct] = zero;

  int lrow = t >> 1, lhalf = t & 1;  // 2 threads per tile row, 32B each
  const __bf16* gA = Xn + (size_t)(m0 + lrow) * N_ + lhalf * 16;
  const __bf16* gB = Wt + (size_t)(c0 + lrow) * N_ + lhalf * 16;
  uint32_t lA[2], lB[2];
  lA[0] = lds_addr32(&As[0][lrow * AP + lhalf * 16]);
  lA[1] = lds_addr32(&As[1][lrow * AP + lhalf * 16]);
  lB[0] = lds_addr32(&Bs[0][lrow * AP + lhalf * 16]);
  lB[1] = lds_addr32(&Bs[1][lrow * AP + lhalf * 16]);

  auto issue = [&](int step, int p) {
    const __bf16* a = gA + (size_t)step * 32;
    const __bf16* b = gB + (size_t)step * 32;
    ASYNC_LD128(lA[p], a, 0);
    ASYNC_LD128(lA[p], a, 16);
    ASYNC_LD128(lB[p], b, 0);
    ASYNC_LD128(lB[p], b, 16);
  };

  const int NSTEP = N_ / 32;  // 128
  issue(0, 0);
  WAIT_ASYNCCNT0();
  __syncthreads();
  for (int i = 0; i < NSTEP; ++i) {
    int p = i & 1;
    if (i + 1 < NSTEP) issue(i + 1, p ^ 1);  // overlaps with WMMA below

    const __bf16* Ab = &As[p][0];
    const __bf16* Bb = &Bs[p][0];
    AF a[2], b[4];
    for (int rt = 0; rt < 2; ++rt) {
      const __bf16* pr = Ab + (rw + 16 * rt + ln) * AP;
      a[rt].u[0] = *(const uint4*)(pr + 8 * hh);
      a[rt].u[1] = *(const uint4*)(pr + 16 + 8 * hh);
    }
    for (int ct = 0; ct < 4; ++ct) {
      const __bf16* pr = Bb + (cw + 16 * ct + ln) * AP;
      b[ct].u[0] = *(const uint4*)(pr + 8 * hh);
      b[ct].u[1] = *(const uint4*)(pr + 16 + 8 * hh);
    }
    for (int rt = 0; rt < 2; ++rt)
      for (int ct = 0; ct < 4; ++ct)
        acc[rt][ct] = __builtin_amdgcn_wmma_f32_16x16x32_bf16(
            false, a[rt].v, false, b[ct].v, (short)0, acc[rt][ct], false, false);

    if (i + 1 < NSTEP) WAIT_ASYNCCNT0();
    __syncthreads();
  }
  for (int rt = 0; rt < 2; ++rt)
    for (int ct = 0; ct < 4; ++ct)
      for (int r = 0; r < 8; ++r) {
        int row = m0 + rw + 16 * rt + r + 8 * hh;
        int col = c0 + cw + 16 * ct + ln;
        Out[(size_t)row * N_ + col] = acc[rt][ct][r];
      }
}

// ----------------------- per-head q/k RMSNorm + KV cache append (+V transp.)
__global__ void __launch_bounds__(128) k_qknorm(const float* __restrict__ Qf,
                                                const float* __restrict__ Kf,
                                                const float* __restrict__ Vf,
                                                __bf16* __restrict__ qn,
                                                __bf16* __restrict__ Kc,
                                                __bf16* __restrict__ Vt) {
  __shared__ float red[128];
  int h = blockIdx.x, m = blockIdx.y, d = threadIdx.x;
  size_t idx = (size_t)m * N_ + h * D_ + d;
  float qv = Qf[idx], kv = Kf[idx], vv = Vf[idx];
  red[d] = qv * qv; __syncthreads();
  for (int s = 64; s > 0; s >>= 1) { if (d < s) red[d] += red[d + s]; __syncthreads(); }
  float rq = rsqrtf(red[0] * (1.0f / D_));
  __syncthreads();
  red[d] = kv * kv; __syncthreads();
  for (int s = 64; s > 0; s >>= 1) { if (d < s) red[d] += red[d + s]; __syncthreads(); }
  float rk = rsqrtf(red[0] * (1.0f / D_));
  qn[idx] = (__bf16)(qv * rq);
  Kc[((size_t)h * L_ + P_ + m) * D_ + d] = (__bf16)(kv * rk);
  Vt[((size_t)h * D_ + d) * L_ + P_ + m] = (__bf16)vv;
}

// ---------------- flash attention, bf16 WMMA core, async double-buffered K/V
#define KP 136  // K-chunk LDS pitch (bf16): 272B rows, 68-dword bank stride
#define VP 40   // V-chunk LDS pitch (bf16)
#define PP 36   // P scratch pitch (f32): 144B rows
__global__ void __launch_bounds__(128) k_attn(const __bf16* __restrict__ qn,
                                              const __bf16* __restrict__ Kc,
                                              const __bf16* __restrict__ Vt,
                                              float* __restrict__ Out) {
  __shared__ __attribute__((aligned(16))) __bf16 KS[2][32 * KP];
  __shared__ __attribute__((aligned(16))) __bf16 VS[2][128 * VP];
  __shared__ __attribute__((aligned(16))) float  PW[4 * 16 * PP];
  int h = blockIdx.y;
  int t = threadIdx.x, lane = t & 31, w = t >> 5;
  int m0 = blockIdx.x * 64 + w * 16;
  int ln = lane & 15, hh = lane >> 4;

  // Q fragments for this wave's 16 rows (resident in VGPRs for whole kernel)
  AF aQ[4];
  {
    const __bf16* qrow = qn + (size_t)(m0 + ln) * N_ + h * D_;
    for (int dk = 0; dk < 4; ++dk) {
      aQ[dk].u[0] = *(const uint4*)(qrow + dk * 32 + 8 * hh);
      aQ[dk].u[1] = *(const uint4*)(qrow + dk * 32 + 16 + 8 * hh);
    }
  }
  v8f zero = {};
  v8f o[8];
  for (int dt = 0; dt < 8; ++dt) o[dt] = zero;
  float mr[8], lr[8];
  for (int r = 0; r < 8; ++r) { mr[r] = -__builtin_inff(); lr[r] = 0.f; }
  float* Pw = PW + w * 16 * PP;

  // staging addresses: each thread moves one 64B row piece of K and of V
  int krow = t >> 2, kq = t & 3;
  const __bf16* gK = Kc + ((size_t)h * L_ + krow) * D_ + kq * 32;  // +c*32*D_
  const __bf16* gV = Vt + ((size_t)h * D_ + t) * L_;               // +c*32
  uint32_t lK[2], lV[2];
  lK[0] = lds_addr32(&KS[0][krow * KP + kq * 32]);
  lK[1] = lds_addr32(&KS[1][krow * KP + kq * 32]);
  lV[0] = lds_addr32(&VS[0][t * VP]);
  lV[1] = lds_addr32(&VS[1][t * VP]);

  auto issue = [&](int c, int p) {
    const __bf16* k = gK + (size_t)c * 32 * D_;
    const __bf16* v = gV + (size_t)c * 32;
    ASYNC_LD128(lK[p], k, 0);
    ASYNC_LD128(lK[p], k, 16);
    ASYNC_LD128(lK[p], k, 32);
    ASYNC_LD128(lK[p], k, 48);
    ASYNC_LD128(lV[p], v, 0);
    ASYNC_LD128(lV[p], v, 16);
    ASYNC_LD128(lV[p], v, 32);
    ASYNC_LD128(lV[p], v, 48);
  };

  const int NCHUNK = L_ / 32;  // 160
  issue(0, 0);
  WAIT_ASYNCCNT0();
  __syncthreads();
  for (int c = 0; c < NCHUNK; ++c) {
    int p = c & 1;
    if (c + 1 < NCHUNK) issue(c + 1, p ^ 1);  // prefetch overlaps everything below

    // S = Q K^T  (two 16x16 n-tiles, K-dim 128 in 4 WMMA steps)
    v8f s0 = zero, s1 = zero;
    const __bf16* Kb = &KS[p][0];
    for (int dk = 0; dk < 4; ++dk) {
      AF b0, b1;
      const __bf16* p0 = Kb + ln * KP + dk * 32 + 8 * hh;
      const __bf16* p1 = Kb + (16 + ln) * KP + dk * 32 + 8 * hh;
      b0.u[0] = *(const uint4*)(p0); b0.u[1] = *(const uint4*)(p0 + 16);
      b1.u[0] = *(const uint4*)(p1); b1.u[1] = *(const uint4*)(p1 + 16);
      s0 = __builtin_amdgcn_wmma_f32_16x16x32_bf16(false, aQ[dk].v, false, b0.v,
                                                   (short)0, s0, false, false);
      s1 = __builtin_amdgcn_wmma_f32_16x16x32_bf16(false, aQ[dk].v, false, b1.v,
                                                   (short)0, s1, false, false);
    }

    // online softmax: rows live at (vgpr r, lane half hh) in both S and O frags
    for (int r = 0; r < 8; ++r) {
      float mx = fmaxf(s0[r], s1[r]);
      mx = fmaxf(mx, __shfl_xor(mx, 1));
      mx = fmaxf(mx, __shfl_xor(mx, 2));
      mx = fmaxf(mx, __shfl_xor(mx, 4));
      mx = fmaxf(mx, __shfl_xor(mx, 8));
      float nm = fmaxf(mr[r], mx);
      float al = __expf(mr[r] - nm);
      mr[r] = nm;
      float a0 = __expf(s0[r] - nm);
      float a1 = __expf(s1[r] - nm);
      float rs = a0 + a1;
      rs += __shfl_xor(rs, 1);
      rs += __shfl_xor(rs, 2);
      rs += __shfl_xor(rs, 4);
      rs += __shfl_xor(rs, 8);
      lr[r] = lr[r] * al + rs;
      for (int dt = 0; dt < 8; ++dt) o[dt][r] *= al;
      Pw[(r + 8 * hh) * PP + ln] = a0;
      Pw[(r + 8 * hh) * PP + 16 + ln] = a1;
    }
    asm volatile("" ::: "memory");  // keep P store->load ordered (LDS in-order per wave)

    // re-layout P: C-frag -> A-frag (through per-wave LDS scratch), cvt to bf16
    AF pa;
    {
      const float* pr = Pw + ln * PP;
      float4 f0 = *(const float4*)(pr + 8 * hh);
      float4 f1 = *(const float4*)(pr + 8 * hh + 4);
      float4 f2 = *(const float4*)(pr + 16 + 8 * hh);
      float4 f3 = *(const float4*)(pr + 16 + 8 * hh + 4);
      pa.v[0] = (__bf16)f0.x;  pa.v[1] = (__bf16)f0.y;
      pa.v[2] = (__bf16)f0.z;  pa.v[3] = (__bf16)f0.w;
      pa.v[4] = (__bf16)f1.x;  pa.v[5] = (__bf16)f1.y;
      pa.v[6] = (__bf16)f1.z;  pa.v[7] = (__bf16)f1.w;
      pa.v[8] = (__bf16)f2.x;  pa.v[9] = (__bf16)f2.y;
      pa.v[10] = (__bf16)f2.z; pa.v[11] = (__bf16)f2.w;
      pa.v[12] = (__bf16)f3.x; pa.v[13] = (__bf16)f3.y;
      pa.v[14] = (__bf16)f3.z; pa.v[15] = (__bf16)f3.w;
    }

    // O += P V  (8 d-tiles)
    const __bf16* Vb = &VS[p][0];
    for (int dt = 0; dt < 8; ++dt) {
      AF bv;
      const __bf16* pv = Vb + (dt * 16 + ln) * VP + 8 * hh;
      bv.u[0] = *(const uint4*)(pv);
      bv.u[1] = *(const uint4*)(pv + 16);
      o[dt] = __builtin_amdgcn_wmma_f32_16x16x32_bf16(false, pa.v, false, bv.v,
                                                      (short)0, o[dt], false, false);
    }

    if (c + 1 < NCHUNK) WAIT_ASYNCCNT0();
    __syncthreads();
  }

  for (int r = 0; r < 8; ++r) {
    float inv = 1.0f / lr[r];
    int row = m0 + r + 8 * hh;
    for (int dt = 0; dt < 8; ++dt)
      Out[(size_t)row * N_ + h * D_ + dt * 16 + ln] = o[dt][r] * inv;
  }
}

// ---------------------------------------------------------------------------
extern "C" void kernel_launch(void* const* d_in, const int* in_sizes, int n_in,
                              void* d_out, int out_size, void* d_ws, size_t ws_size,
                              hipStream_t stream) {
  (void)in_sizes; (void)n_in; (void)out_size; (void)ws_size;
  const float* X  = (const float*)d_in[0];
  const float* Wq = (const float*)d_in[1];
  const float* Wk = (const float*)d_in[2];
  const float* Wv = (const float*)d_in[3];
  const float* cK = (const float*)d_in[4];
  const float* cV = (const float*)d_in[5];
  float* Out = (float*)d_out;

  char* ws = (char*)d_ws;
  size_t off = 0;
  __bf16* Xn = (__bf16*)(ws + off); off += (size_t)M_ * N_ * 2;        // 8 MB
  __bf16* Wt = (__bf16*)(ws + off); off += (size_t)3 * N_ * N_ * 2;    // 96 MB
  float* Qf = (float*)(ws + off);   off += (size_t)M_ * N_ * 4;        // 16 MB
  float* Kf = (float*)(ws + off);   off += (size_t)M_ * N_ * 4;        // 16 MB
  float* Vf = (float*)(ws + off);   off += (size_t)M_ * N_ * 4;        // 16 MB
  __bf16* qn = (__bf16*)(ws + off); off += (size_t)M_ * N_ * 2;        // 8 MB
  __bf16* Kc = (__bf16*)(ws + off); off += (size_t)H_ * L_ * D_ * 2;   // 40 MB
  __bf16* Vt = (__bf16*)(ws + off); off += (size_t)H_ * L_ * D_ * 2;   // 40 MB

  k_rmsnorm_x<<<M_, 256, 0, stream>>>(X, Xn);
  k_wt<<<dim3(N_ / 32, N_ / 32, 3), 256, 0, stream>>>(Wq, Wk, Wv, Wt);
  k_cachek<<<2048, 256, 0, stream>>>(cK, Kc);
  k_cachev<<<dim3(P_ / 32, D_ / 32, H_), 256, 0, stream>>>(cV, Vt);
  k_gemm<<<dim3(N_ / 128, M_ / 128, 3), 256, 0, stream>>>(Xn, Wt, Qf, Kf, Vf);
  k_qknorm<<<dim3(H_, M_), 128, 0, stream>>>(Qf, Kf, Vf, qn, Kc, Vt);
  k_attn<<<dim3(M_ / 64, H_), 128, 0, stream>>>(qn, Kc, Vt, Out);
}
